// SelKDLoss_53626961658396
// MI455X (gfx1250) — compile-verified, matrix operand order checked by reading, not run on hardware
//
#include <hip/hip_runtime.h>
#include <hip/hip_bf16.h>

// Problem constants (from reference setup_inputs)
#define BB 16
#define HH 16
#define MM 256
#define NN 576
#define DD 1024
#define FI 0.9090909090909091f   // RHO/(RHO+EPSILON) = 1/1.1
#define EPSK 1e-8f

typedef __attribute__((ext_vector_type(16))) __bf16 v16bf;
typedef __attribute__((ext_vector_type(8)))  __bf16 v8bf;
typedef __attribute__((ext_vector_type(8)))  float  v8f;

union BFrag { v16bf v; v8bf h[2]; };

// ---------------- block reduction (wave32) ----------------
__device__ __forceinline__ float block_reduce_sum(float val, float* smem) {
    const int lane = threadIdx.x & 31;
    const int wid  = threadIdx.x >> 5;
    #pragma unroll
    for (int off = 16; off; off >>= 1) val += __shfl_xor(val, off, 32);
    if (lane == 0) smem[wid] = val;
    __syncthreads();
    const int nwaves = blockDim.x >> 5;
    val = (threadIdx.x < nwaves) ? smem[threadIdx.x] : 0.0f;
    if (wid == 0) {
        #pragma unroll
        for (int off = 16; off; off >>= 1) val += __shfl_xor(val, off, 32);
    }
    return val;  // valid in thread 0
}

// ---------------- init: v = 1, d_out = 0 ----------------
__global__ void init_kernel(float* __restrict__ v, float* __restrict__ out) {
    int i = blockIdx.x * blockDim.x + threadIdx.x;
    if (i < BB * NN) v[i] = 1.0f;
    if (i == 0) out[0] = 0.0f;
}

// ---------------- per-batch mask sum ----------------
__global__ void masksum_kernel(const float* __restrict__ mask, float* __restrict__ msum) {
    __shared__ float smem[8];
    int b = blockIdx.x;
    float val = mask[b * MM + threadIdx.x];      // blockDim.x == MM == 256
    val = block_reduce_sum(val, smem);
    if (threadIdx.x == 0) msum[b] = val;
}

// ---------------- row L2-normalize fp32 -> bf16 ----------------
__global__ void l2norm_bf16_kernel(const float* __restrict__ in, __bf16* __restrict__ out) {
    __shared__ float smem[8];
    __shared__ float rn_s;
    const size_t row = blockIdx.x;
    const float* r = in + row * DD;
    __bf16* o = out + row * DD;
    float ss = 0.0f;
    for (int i = threadIdx.x; i < (DD >> 2); i += blockDim.x) {
        float4 v = reinterpret_cast<const float4*>(r)[i];
        ss += v.x * v.x + v.y * v.y + v.z * v.z + v.w * v.w;
    }
    ss = block_reduce_sum(ss, smem);
    if (threadIdx.x == 0) rn_s = rsqrtf(ss + 1e-12f);
    __syncthreads();
    const float s = rn_s;
    for (int i = threadIdx.x; i < (DD >> 2); i += blockDim.x) {
        float4 v = reinterpret_cast<const float4*>(r)[i];
        int j = i << 2;
        o[j + 0] = (__bf16)(v.x * s);
        o[j + 1] = (__bf16)(v.y * s);
        o[j + 2] = (__bf16)(v.z * s);
        o[j + 3] = (__bf16)(v.w * s);
    }
}

// ---------------- WMMA cost GEMM + exp -> K ----------------
// K[b,m,n] = exp(-(1 - yn[b,m,:].xn[b,n,:]) / 0.1) = exp(10*dot - 10)
// One wave per 16x16 output tile; K-dim (D=1024) in steps of 32 bf16.
__global__ void __launch_bounds__(128)
cost_gemm_kernel(const __bf16* __restrict__ yn, const __bf16* __restrict__ xn,
                 float* __restrict__ Kmat) {
    const int TM = MM / 16;           // 16
    const int TN = NN / 16;           // 36
    const int wave = blockIdx.x * 4 + (threadIdx.x >> 5);
    const int lane = threadIdx.x & 31;
    int b  = wave / (TM * TN);
    int r  = wave % (TM * TN);
    int mt = r / TN;
    int nt = r % TN;
    const int half = lane >> 4;       // 0: lanes 0-15, 1: lanes 16-31
    const int l16  = lane & 15;

    // A: 16x32 bf16 tile of yn (rows m), B: 32x16 tile fed from xn rows (n)
    const __bf16* arow = yn + (size_t)(b * MM + mt * 16 + l16) * DD;
    const __bf16* brow = xn + (size_t)(b * NN + nt * 16 + l16) * DD;

    v8f acc = {};
    for (int k0 = 0; k0 < DD; k0 += 32) {
        const int klo = k0 + half * 8;   // ISA 16-bit A layout: lanes hold two 8-elem chunks
        BFrag a, bm;
        a.h[0]  = *reinterpret_cast<const v8bf*>(arow + klo);
        a.h[1]  = *reinterpret_cast<const v8bf*>(arow + klo + 16);
        bm.h[0] = *reinterpret_cast<const v8bf*>(brow + klo);
        bm.h[1] = *reinterpret_cast<const v8bf*>(brow + klo + 16);
        acc = __builtin_amdgcn_wmma_f32_16x16x32_bf16(
                  false, a.v, false, bm.v, (short)0, acc, false, false);
    }

    // C/D layout: VGPR r -> row (r + 8*half), col = l16
    const int n = nt * 16 + l16;
    float* out = Kmat + ((size_t)(b * MM + mt * 16 + half * 8)) * NN + n;
    #pragma unroll
    for (int rr = 0; rr < 8; ++rr) {
        out[(size_t)rr * NN] = __expf(10.0f * acc[rr] - 10.0f);
    }
}

// ---------------- Sinkhorn: u update ----------------
__global__ void sinkhorn_u_kernel(const float* __restrict__ Kmat, const float* __restrict__ v,
                                  const float* __restrict__ mask, const float* __restrict__ msum,
                                  float* __restrict__ u) {
    __shared__ float smem[8];
    const int row = blockIdx.x;              // b*MM + m
    const int b = row >> 8;
    const float* kr = Kmat + (size_t)row * NN;
    const float* vb = v + b * NN;
    float acc = 0.0f;
    for (int n = threadIdx.x; n < NN; n += blockDim.x) acc += kr[n] * vb[n];
    acc = block_reduce_sum(acc, smem);
    if (threadIdx.x == 0) {
        float nu = mask[row] / (msum[b] + EPSK);
        u[row] = __powf(nu / (acc + EPSK), FI);
    }
}

// ---------------- Sinkhorn: v update ----------------
__global__ void sinkhorn_v_kernel(const float* __restrict__ Kmat, const float* __restrict__ u,
                                  float* __restrict__ v) {
    __shared__ float smem[8];
    const int col = blockIdx.x;              // b*NN + n
    const int b = col / NN;
    const int n = col - b * NN;
    const float* kb = Kmat + (size_t)b * MM * NN + n;
    const float* ub = u + b * MM;
    float acc = 0.0f;
    for (int m = threadIdx.x; m < MM; m += blockDim.x) acc += kb[(size_t)m * NN] * ub[m];
    acc = block_reduce_sum(acc, smem);
    if (threadIdx.x == 0) {
        v[col] = __powf((1.0f / (float)NN) / (acc + EPSK), FI);
    }
}

// ---------------- teacher = row-normalized u*K*v (in-place on K) ----------------
__global__ void teacher_kernel(float* __restrict__ Kmat, const float* __restrict__ u,
                               const float* __restrict__ v) {
    __shared__ float smem[8];
    __shared__ float inv_s;
    const int row = blockIdx.x;
    const int b = row >> 8;
    float* kr = Kmat + (size_t)row * NN;
    const float* vb = v + b * NN;
    const float uu = u[row];
    float p[3];
    float acc = 0.0f;
    int idx = 0;
    for (int n = threadIdx.x; n < NN; n += blockDim.x) {
        p[idx] = uu * kr[n] * vb[n];
        acc += p[idx];
        ++idx;
    }
    acc = block_reduce_sum(acc, smem);
    if (threadIdx.x == 0) inv_s = 1.0f / (acc + EPSK);
    __syncthreads();
    const float s = inv_s;
    idx = 0;
    for (int n = threadIdx.x; n < NN; n += blockDim.x) {
        kr[n] = p[idx] * s;
        ++idx;
    }
}

// ---------------- streaming loss: 151 MB of student_attn, HBM-bound ----------------
// One block per (b,m): teacher row cached in LDS, reused across all 16 heads.
__global__ void __launch_bounds__(256)
loss_kernel(const float* __restrict__ teacher, const float* __restrict__ student,
            const float* __restrict__ mask, float* __restrict__ out) {
    __shared__ float trow[NN];
    __shared__ float smem[8];
    const int row = blockIdx.x;              // b*MM + m
    const int b = row >> 8;
    const int m = row & (MM - 1);
    const float* tr = teacher + (size_t)row * NN;
    for (int n = threadIdx.x; n < NN; n += blockDim.x) trow[n] = tr[n];
    __syncthreads();

    const float* sb = student + (((size_t)b * HH) * MM + m) * NN; // h = 0
    const size_t hstride = (size_t)MM * NN;
    float acc = 0.0f;
    // flatten (h, n/4): 16 * 144 = 2304 float4 per block
    for (int i = threadIdx.x; i < HH * (NN >> 2); i += blockDim.x) {
        const int h  = i / (NN >> 2);
        const int n4 = i - h * (NN >> 2);
        const float4 s4 = reinterpret_cast<const float4*>(sb + h * hstride)[n4];
        const int n = n4 << 2;
        acc += trow[n + 0] * __logf(s4.x + EPSK)
             + trow[n + 1] * __logf(s4.y + EPSK)
             + trow[n + 2] * __logf(s4.z + EPSK)
             + trow[n + 3] * __logf(s4.w + EPSK);
    }
    acc = block_reduce_sum(acc, smem);
    if (threadIdx.x == 0) {
        // mean over B*H*M = 65536 rows, negated
        atomicAdd(out, -acc * mask[row] * (1.0f / (float)(BB * HH * MM)));
    }
}

extern "C" void kernel_launch(void* const* d_in, const int* in_sizes, int n_in,
                              void* d_out, int out_size, void* d_ws, size_t ws_size,
                              hipStream_t stream) {
    const float* student = (const float*)d_in[0];   // (B,H,M,N)
    const float* visual  = (const float*)d_in[1];   // (B,N,D)
    const float* text    = (const float*)d_in[2];   // (B,M,D)
    const float* mask    = (const float*)d_in[3];   // (B,M)
    float* out = (float*)d_out;

    // workspace layout
    char* ws = (char*)d_ws;
    __bf16* xn = (__bf16*)ws;                               // B*N*D bf16 = 18,874,368 B
    size_t off = (size_t)BB * NN * DD * 2;
    __bf16* yn = (__bf16*)(ws + off);                       // B*M*D bf16 =  8,388,608 B
    off += (size_t)BB * MM * DD * 2;
    float* Kmat = (float*)(ws + off);                       // B*M*N f32  =  9,437,184 B
    off += (size_t)BB * MM * NN * 4;
    float* u = (float*)(ws + off);  off += (size_t)BB * MM * 4;
    float* v = (float*)(ws + off);  off += (size_t)BB * NN * 4;
    float* msum = (float*)(ws + off);                       // total ~36.8 MB

    // 1. init v=1, zero output accumulator
    init_kernel<<<(BB * NN + 255) / 256, 256, 0, stream>>>(v, out);

    // 2. per-batch mask sums
    masksum_kernel<<<BB, MM, 0, stream>>>(mask, msum);

    // 3. L2-normalize features, cast to bf16
    l2norm_bf16_kernel<<<BB * NN, 256, 0, stream>>>(visual, xn);
    l2norm_bf16_kernel<<<BB * MM, 256, 0, stream>>>(text, yn);

    // 4. WMMA cost GEMM -> K = exp(10*dot - 10). 1 tile/wave, 4 waves/block.
    {
        const int tiles = BB * (MM / 16) * (NN / 16);       // 9216
        cost_gemm_kernel<<<tiles / 4, 128, 0, stream>>>(yn, xn, Kmat);
    }

    // 5. Sinkhorn iterations
    for (int it = 0; it < 5; ++it) {
        sinkhorn_u_kernel<<<BB * MM, 256, 0, stream>>>(Kmat, v, mask, msum, u);
        sinkhorn_v_kernel<<<BB * NN, 256, 0, stream>>>(Kmat, u, v);
    }

    // 6. teacher plan, row-normalized, overwrites K in-place
    teacher_kernel<<<BB * MM, 256, 0, stream>>>(Kmat, u, v);

    // 7. streaming cross-entropy loss (HBM-bound: 151 MB student reads)
    loss_kernel<<<BB * MM, 256, 0, stream>>>(Kmat, student, mask, out);
}
